// PatternLearner_28501402976355
// MI455X (gfx1250) — matrix operand order repaired and checked by reading, not compiled, not run on hardware
//
#include <hip/hip_runtime.h>
#include <math.h>

typedef __bf16 bf16;
typedef __attribute__((ext_vector_type(16))) __bf16 v16bf;
typedef __attribute__((ext_vector_type(8)))  __bf16 v8bf;
typedef __attribute__((ext_vector_type(8)))  float  v8f;

#define NTOK  8192      // B*S
#define HDIM  768
#define H2DIM 1536
#define PDIM  8192
#define KNN   16
#define KCH   (HDIM / 32)   // 24 K-chunks of 32 in the dist GEMM
#define EPSC  1e-5f

// ---------------- CDNA5 async LDS copy helpers ----------------
// global_load_async_to_lds_b128: per-lane 16B global -> LDS, tracked by ASYNCcnt
__device__ __forceinline__ void async_g2lds_b128(void* lds_dst, const void* gsrc) {
  unsigned int lds_addr = (unsigned int)(unsigned long long)(size_t)lds_dst;
  unsigned long long gaddr = (unsigned long long)(size_t)gsrc;
  asm volatile("global_load_async_to_lds_b128 %0, %1, off"
               :: "v"(lds_addr), "v"(gaddr)
               : "memory");
}
__device__ __forceinline__ void wait_asynccnt0() {
  asm volatile("s_wait_asynccnt 0x0" ::: "memory");
}

// ---------------- helpers ----------------

__device__ __forceinline__ float block_reduce_sum_256(float v, float* red) {
  const int t = threadIdx.x;
  red[t] = v;
  __syncthreads();
  #pragma unroll
  for (int s = 128; s > 0; s >>= 1) {
    if (t < s) red[t] += red[t + s];
    __syncthreads();
  }
  float r = red[0];
  __syncthreads();
  return r;
}

__device__ __forceinline__ v16bf cat8(v8bf lo, v8bf hi) {
  return __builtin_shufflevector(lo, hi, 0,1,2,3,4,5,6,7,8,9,10,11,12,13,14,15);
}

// ---------------- conversion kernels ----------------

// Wt[n*K + k] = (bf16) W[k*N + n]   (weights become N-major so WMMA B-fragments
// are two contiguous b128 loads per lane, no LDS needed for B)
__global__ __launch_bounds__(256) void cvt_transpose_bf16(const float* __restrict__ W,
                                                          bf16* __restrict__ Wt,
                                                          int K, int N) {
  int idx = blockIdx.x * 256 + threadIdx.x;
  if (idx < K * N) {
    int k = idx / N, n = idx % N;           // reads coalesced along n
    Wt[(size_t)n * K + k] = (bf16)W[idx];
  }
}

// pattern memory prep: bf16 copy, clipped squared norm, sigmoid(importance)
__global__ __launch_bounds__(256) void mem_prep(const float* __restrict__ M,
                                                const float* __restrict__ imp,
                                                bf16* __restrict__ Mbf,
                                                float* __restrict__ B2,
                                                float* __restrict__ Sig) {
  __shared__ float red[256];
  const int row = blockIdx.x;
  const float* m = M + (size_t)row * HDIM;
  bf16* o = Mbf + (size_t)row * HDIM;
  float ss = 0.f;
  for (int i = threadIdx.x; i < HDIM; i += 256) {
    float v = m[i];
    ss += v * v;
    o[i] = (bf16)v;
  }
  float tot = block_reduce_sum_256(ss, red);
  if (threadIdx.x == 0) {
    B2[row] = fminf(fmaxf(tot, 0.f), 1.f - EPSC);
    Sig[row] = 1.f / (1.f + expf(-imp[row]));
  }
}

// ---------------- hyperbolic maps ----------------

__global__ __launch_bounds__(256) void log0_rows(const float* __restrict__ X,
                                                 bf16* __restrict__ O) {
  __shared__ float red[256];
  const int row = blockIdx.x;
  const float* x = X + (size_t)row * HDIM;
  float ss = 0.f;
  for (int i = threadIdx.x; i < HDIM; i += 256) { float v = x[i]; ss += v * v; }
  float tot = block_reduce_sum_256(ss, red);
  float nrm = sqrtf(tot);
  float nc = fminf(fmaxf(nrm, EPSC), 1.f - EPSC);
  float scale = atanhf(nc) / nc;
  bf16* o = O + (size_t)row * HDIM;
  for (int i = threadIdx.x; i < HDIM; i += 256) o[i] = (bf16)(x[i] * scale);
}

// exp0 producing: pattern fp32 (to d_out), pattern bf16 (for dist GEMM), clipped |a|^2
__global__ __launch_bounds__(256) void exp0_pattern(const float* __restrict__ Hin,
                                                    float* __restrict__ PatOut,
                                                    bf16* __restrict__ Pbf,
                                                    float* __restrict__ A2) {
  __shared__ float red[256];
  const int row = blockIdx.x;
  const float* h = Hin + (size_t)row * HDIM;
  float ss = 0.f;
  for (int i = threadIdx.x; i < HDIM; i += 256) { float v = h[i]; ss += v * v; }
  float tot = block_reduce_sum_256(ss, red);
  float n = fmaxf(sqrtf(tot), EPSC);
  float t = tanhf(n);
  float sc = t / n;
  float* po = PatOut + (size_t)row * HDIM;
  bf16*  pb = Pbf + (size_t)row * HDIM;
  for (int i = threadIdx.x; i < HDIM; i += 256) {
    float p = h[i] * sc;
    po[i] = p;
    pb[i] = (bf16)p;
  }
  if (threadIdx.x == 0)
    A2[row] = fminf(fmaxf(sc * sc * tot, 0.f), 1.f - EPSC);
}

__global__ __launch_bounds__(256) void exp0_out(const float* __restrict__ Hin,
                                                float* __restrict__ Out) {
  __shared__ float red[256];
  const int row = blockIdx.x;
  const float* h = Hin + (size_t)row * HDIM;
  float ss = 0.f;
  for (int i = threadIdx.x; i < HDIM; i += 256) { float v = h[i]; ss += v * v; }
  float tot = block_reduce_sum_256(ss, red);
  float n = fmaxf(sqrtf(tot), EPSC);
  float sc = tanhf(n) / n;
  float* o = Out + (size_t)row * HDIM;
  for (int i = threadIdx.x; i < HDIM; i += 256) o[i] = h[i] * sc;
}

// ---------------- LayerNorm + exact GELU ----------------

__global__ __launch_bounds__(256) void ln_gelu_rows(const float* __restrict__ Hin,
                                                    const float* __restrict__ g,
                                                    const float* __restrict__ be,
                                                    bf16* __restrict__ O, int W) {
  __shared__ float red[256];
  const int row = blockIdx.x;
  const float* h = Hin + (size_t)row * W;
  float s = 0.f;
  for (int i = threadIdx.x; i < W; i += 256) s += h[i];
  float mu = block_reduce_sum_256(s, red) / (float)W;
  float vs = 0.f;
  for (int i = threadIdx.x; i < W; i += 256) { float d = h[i] - mu; vs += d * d; }
  float var = block_reduce_sum_256(vs, red) / (float)W;
  float inv = rsqrtf(var + 1e-5f);
  bf16* o = O + (size_t)row * W;
  for (int i = threadIdx.x; i < W; i += 256) {
    float y = (h[i] - mu) * inv * g[i] + be[i];
    float gel = 0.5f * y * (1.f + erff(y * 0.70710678118654752f));
    o[i] = (bf16)gel;
  }
}

// ---------------- WMMA GEMM: C[M,N] = A[M,K] @ Wt^T + bias ----------------
// A[M,K] bf16 row-major; Wt[N,K] bf16 row-major (pre-transposed weights).
// 256 threads = 8 waves (2 row-groups x 4 col-groups); block tile 32x64.
// A staged 64-wide-K, DOUBLE-BUFFERED via CDNA5 async global->LDS b128 so the
// copy of tile k+1 overlaps the two WMMAs of tile k; 1 barrier + 1 wait / 2 WMMA.
// B fragments load straight from global as two contiguous b128 per lane.

__global__ __launch_bounds__(256) void gemm_bias_bf16(const bf16* __restrict__ A,
                                                      const bf16* __restrict__ Wt,
                                                      const float* __restrict__ bias,
                                                      float* __restrict__ C,
                                                      int M, int N, int K) {
  __shared__ bf16 As[2][32][72];  // 64-wide K tile + pad 8 -> 144B row stride
  const int tid  = threadIdx.x;
  const int lane = tid & 31;
  const int wave = tid >> 5;
  const int wm = wave >> 2;          // 0..1
  const int wn = wave & 3;           // 0..3
  const int n16  = lane & 15;
  const int kgrp = lane >> 4;
  const int row0 = blockIdx.y * 32;
  const int col0 = blockIdx.x * 64;

  const int col = col0 + wn * 16 + n16;
  const bf16* bp = Wt + (size_t)col * K;   // this lane's B column, contiguous in K

  // each thread stages one b128 (8 bf16) of the 32x64 A tile
  const int sr = tid >> 3;           // 0..31
  const int sc = (tid & 7) * 8;      // 0..56
  const bf16* arow = A + (size_t)(row0 + sr) * K + sc;

  v8f acc = {};

  // prologue: stage tile 0 into buffer 0
  async_g2lds_b128(&As[0][sr][sc], arow);

  int buf = 0;
  for (int k0 = 0; k0 < K; k0 += 64, buf ^= 1) {
    wait_asynccnt0();
    __syncthreads();
    if (k0 + 64 < K)
      async_g2lds_b128(&As[buf ^ 1][sr][sc], arow + k0 + 64);

    #pragma unroll
    for (int s = 0; s < 2; ++s) {
      const int ks = s * 32;
      // A fragment (16-bit A 16x32 layout)
      v8bf alo = *(const v8bf*)&As[buf][wm * 16 + n16][ks + kgrp * 8];
      v8bf ahi = *(const v8bf*)&As[buf][wm * 16 + n16][ks + 16 + kgrp * 8];
      // B fragment (same K order as A): lanes 0-15 K=0..7,16..23, lanes 16-31 K=8..15,24..31
      v8bf blo = *(const v8bf*)(bp + k0 + ks + kgrp * 8);
      v8bf bhi = *(const v8bf*)(bp + k0 + ks + 16 + kgrp * 8);
      acc = __builtin_amdgcn_wmma_f32_16x16x32_bf16(false, cat8(alo, ahi),
                                                    false, cat8(blo, bhi),
                                                    (short)0, acc, false, false);
    }
  }

  // C/D layout: VGPR j -> row m = j + 8*kgrp, col = lane&15
  const float bb = bias[col];
  #pragma unroll
  for (int j = 0; j < 8; ++j) {
    const int row = row0 + wm * 16 + j + 8 * kgrp;
    C[(size_t)row * N + col] = acc[j] + bb;
  }
}

// ---------------- fused Poincare distance + top-16 + weighted gather ----------------
// 128 threads = 4 independent waves, each owns 16 token rows.
// A fragments for the wave's 16 rows are REGISTER-RESIDENT (24 x v16bf = 192 VGPRs),
// so the hot loop (512 P-tiles x 24 WMMAs) is just 2 global b128 loads + 1 WMMA.

__global__ __launch_bounds__(128) void dist_topk_gather(const bf16* __restrict__ Abf,
                                                        const float* __restrict__ A2,
                                                        const bf16* __restrict__ Mbf,
                                                        const float* __restrict__ B2,
                                                        const float* __restrict__ Sig,
                                                        const float* __restrict__ Mf,
                                                        float* __restrict__ Proc) {
  __shared__ float dtile[4][16][17];
  __shared__ float topd[4][16][KNN];
  __shared__ int   topi[4][16][KNN];
  __shared__ float a2s[4][16];

  const int tid  = threadIdx.x;
  const int lane = tid & 31;
  const int wave = tid >> 5;
  const int n16  = lane & 15;
  const int kgrp = lane >> 4;
  const int r0 = (blockIdx.x * 4 + wave) * 16;

  // preload this wave's A fragments (row m = n16, K split per kgrp) into registers
  v16bf afr[KCH];
  {
    const bf16* arow = Abf + (size_t)(r0 + n16) * HDIM;
    #pragma unroll
    for (int kk = 0; kk < KCH; ++kk) {
      v8bf alo = *(const v8bf*)(arow + kk * 32 + kgrp * 8);
      v8bf ahi = *(const v8bf*)(arow + kk * 32 + 16 + kgrp * 8);
      afr[kk] = cat8(alo, ahi);
    }
  }

  for (int idx = lane; idx < 16 * KNN; idx += 32) {
    topd[wave][idx >> 4][idx & 15] = 3.0e38f;
    topi[wave][idx >> 4][idx & 15] = 0;
  }
  if (lane < 16) a2s[wave][lane] = A2[r0 + lane];
  __syncthreads();

  #pragma unroll 1
  for (int p0 = 0; p0 < PDIM; p0 += 16) {
    v8f acc = {};
    const bf16* bp = &Mbf[(size_t)(p0 + n16) * HDIM];
    #pragma unroll
    for (int kk = 0; kk < KCH; ++kk) {
      v8bf blo = *(const v8bf*)(bp + kk * 32 + kgrp * 8);
      v8bf bhi = *(const v8bf*)(bp + kk * 32 + 16 + kgrp * 8);
      acc = __builtin_amdgcn_wmma_f32_16x16x32_bf16(false, afr[kk],
                                                    false, cat8(blo, bhi),
                                                    (short)0, acc, false, false);
    }
    // epilogue: Poincare metric * sigmoid(importance)
    {
      const int p = p0 + n16;
      const float b2 = B2[p];
      const float sg = Sig[p];
      #pragma unroll
      for (int j = 0; j < 8; ++j) {
        const int m = j + 8 * kgrp;
        const float a2 = a2s[wave][m];
        float num = fmaxf(a2 + b2 - 2.f * acc[j], 0.f);
        float arg = 1.f + 2.f * num / ((1.f - a2) * (1.f - b2));
        arg = fmaxf(arg, 1.f + 1e-7f);
        dtile[wave][m][n16] = acoshf(arg) * sg;
      }
    }
    __syncthreads();
    // running top-16 smallest per row (sorted ascending insertion)
    if (lane < 16) {
      const int m = lane;
      float worst = topd[wave][m][KNN - 1];
      for (int n = 0; n < 16; ++n) {
        float v = dtile[wave][m][n];
        if (v < worst) {
          int pos = KNN - 1;
          while (pos > 0 && topd[wave][m][pos - 1] > v) {
            topd[wave][m][pos] = topd[wave][m][pos - 1];
            topi[wave][m][pos] = topi[wave][m][pos - 1];
            --pos;
          }
          topd[wave][m][pos] = v;
          topi[wave][m][pos] = p0 + n;
          worst = topd[wave][m][KNN - 1];
        }
      }
    }
    __syncthreads();
  }

  // weighted gather: processed[row] = sum_k exp(-d_k) * mem[idx_k]
  for (int m = 0; m < 16; ++m) {
    const int row = r0 + m;
    for (int h = lane; h < HDIM; h += 32) {
      float s = 0.f;
      #pragma unroll
      for (int k = 0; k < KNN; ++k) {
        float w = expf(-topd[wave][m][k]);
        s += w * Mf[(size_t)topi[wave][m][k] * HDIM + h];
      }
      Proc[(size_t)row * HDIM + h] = s;
    }
  }
}

// ---------------- orchestration ----------------

extern "C" void kernel_launch(void* const* d_in, const int* in_sizes, int n_in,
                              void* d_out, int out_size, void* d_ws, size_t ws_size,
                              hipStream_t stream) {
  const float* x       = (const float*)d_in[0];
  const float* enc_w1  = (const float*)d_in[1];
  const float* enc_b1  = (const float*)d_in[2];
  const float* enc_g   = (const float*)d_in[3];
  const float* enc_be  = (const float*)d_in[4];
  const float* enc_w2  = (const float*)d_in[5];
  const float* enc_b2  = (const float*)d_in[6];
  const float* dec_w1  = (const float*)d_in[7];
  const float* dec_b1  = (const float*)d_in[8];
  const float* dec_g   = (const float*)d_in[9];
  const float* dec_be  = (const float*)d_in[10];
  const float* dec_w2  = (const float*)d_in[11];
  const float* dec_b2  = (const float*)d_in[12];
  const float* mem     = (const float*)d_in[13];
  const float* imp     = (const float*)d_in[14];

  float* out     = (float*)d_out;                       // [N,H] output
  float* pat_out = out + (size_t)NTOK * HDIM;           // [N,H] pattern (tuple 2nd)

  char* base = (char*)d_ws;
  size_t off = 0;
  auto alloc = [&](size_t bytes) {
    char* p = base + off;
    off += (bytes + 255) & ~(size_t)255;
    return p;
  };
  bf16*  wew1 = (bf16*) alloc((size_t)HDIM * H2DIM * 2);  // [H2DIM, HDIM] = enc_w1^T
  bf16*  wew2 = (bf16*) alloc((size_t)H2DIM * HDIM * 2);  // [HDIM, H2DIM] = enc_w2^T
  bf16*  wdw1 = (bf16*) alloc((size_t)HDIM * H2DIM * 2);
  bf16*  wdw2 = (bf16*) alloc((size_t)H2DIM * HDIM * 2);
  bf16*  xlog = (bf16*) alloc((size_t)NTOK * HDIM * 2);
  float* hbuf = (float*)alloc((size_t)NTOK * H2DIM * 4);
  bf16*  gbuf = (bf16*) alloc((size_t)NTOK * H2DIM * 2);
  bf16*  pbf  = (bf16*) alloc((size_t)NTOK * HDIM * 2);
  bf16*  mbf  = (bf16*) alloc((size_t)PDIM * HDIM * 2);
  float* a2   = (float*)alloc((size_t)NTOK * 4);
  float* b2   = (float*)alloc((size_t)PDIM * 4);
  float* sig  = (float*)alloc((size_t)PDIM * 4);
  float* proc = (float*)alloc((size_t)NTOK * HDIM * 4);
  (void)ws_size; (void)n_in; (void)in_sizes; (void)out_size;

  const int WELEM = HDIM * H2DIM;
  const int WBLK = (WELEM + 255) / 256;
  cvt_transpose_bf16<<<WBLK, 256, 0, stream>>>(enc_w1, wew1, HDIM, H2DIM);
  cvt_transpose_bf16<<<WBLK, 256, 0, stream>>>(enc_w2, wew2, H2DIM, HDIM);
  cvt_transpose_bf16<<<WBLK, 256, 0, stream>>>(dec_w1, wdw1, HDIM, H2DIM);
  cvt_transpose_bf16<<<WBLK, 256, 0, stream>>>(dec_w2, wdw2, H2DIM, HDIM);
  mem_prep<<<PDIM, 256, 0, stream>>>(mem, imp, mbf, b2, sig);

  // encoder
  log0_rows<<<NTOK, 256, 0, stream>>>(x, xlog);
  gemm_bias_bf16<<<dim3(H2DIM / 64, NTOK / 32), 256, 0, stream>>>(
      xlog, wew1, enc_b1, hbuf, NTOK, H2DIM, HDIM);
  ln_gelu_rows<<<NTOK, 256, 0, stream>>>(hbuf, enc_g, enc_be, gbuf, H2DIM);
  gemm_bias_bf16<<<dim3(HDIM / 64, NTOK / 32), 256, 0, stream>>>(
      gbuf, wew2, enc_b2, hbuf, NTOK, HDIM, H2DIM);
  exp0_pattern<<<NTOK, 256, 0, stream>>>(hbuf, pat_out, pbf, a2);

  // fused hyperbolic kNN retrieval
  dist_topk_gather<<<NTOK / 64, 128, 0, stream>>>(pbf, a2, mbf, b2, sig, mem, proc);

  // decoder
  log0_rows<<<NTOK, 256, 0, stream>>>(proc, xlog);
  gemm_bias_bf16<<<dim3(H2DIM / 64, NTOK / 32), 256, 0, stream>>>(
      xlog, wdw1, dec_b1, hbuf, NTOK, H2DIM, HDIM);
  ln_gelu_rows<<<NTOK, 256, 0, stream>>>(hbuf, dec_g, dec_be, gbuf, H2DIM);
  gemm_bias_bf16<<<dim3(HDIM / 64, NTOK / 32), 256, 0, stream>>>(
      gbuf, wdw2, dec_b2, hbuf, NTOK, HDIM, H2DIM);
  exp0_out<<<NTOK, 256, 0, stream>>>(hbuf, out);
}
